// LayoutGraphModel_39616778338829
// MI455X (gfx1250) — matrix operand order repaired and checked by reading, not compiled, not run on hardware
//
#include <hip/hip_runtime.h>

// ---------------------------------------------------------------------------
// Types for CDNA5 WMMA (wave32): v_wmma_f32_16x16x32_bf16
// ---------------------------------------------------------------------------
typedef __attribute__((ext_vector_type(16))) __bf16        v16bf;
typedef __attribute__((ext_vector_type(8)))  float         v8f;
typedef __attribute__((ext_vector_type(4)))  unsigned int  v4u;

union Frag16 { v16bf v; v4u q[2]; };   // 32 bytes: one 16x32 A tile or 32x16 B tile per lane

__device__ __forceinline__ unsigned short f2bf(float f) {
  unsigned int u = __float_as_uint(f);
  u += 0x7FFFu + ((u >> 16) & 1u);     // round-to-nearest-even
  return (unsigned short)(u >> 16);
}
__device__ __forceinline__ float bf2f(unsigned short h) {
  return __uint_as_float(((unsigned int)h) << 16);
}
// searchsorted(sep, row, side='right')
__device__ __forceinline__ int seg_of(int row, const int* __restrict__ sep, int nseg) {
  int lo = 0, hi = nseg - 1;
  while (lo < hi) { int mid = (lo + hi) >> 1; if (sep[mid] <= row) lo = mid + 1; else hi = mid; }
  return lo;
}
__device__ __forceinline__ float leaky(float v) { return v > 0.f ? v : 0.01f * v; }

// ---------------------------------------------------------------------------
// Utility kernels
// ---------------------------------------------------------------------------
__global__ void zerof_kernel(float* __restrict__ p, int n) {
  for (int i = blockIdx.x * blockDim.x + threadIdx.x; i < n; i += gridDim.x * blockDim.x)
    p[i] = 0.f;
}

// fp32 [Mreal,Ks] -> bf16 [M,Kd] (zero padded)
__global__ void conv_w_kernel(const float* __restrict__ src, unsigned short* __restrict__ dst,
                              int M, int Mreal, int Ks, int Kd) {
  int total = M * Kd;
  for (int i = blockIdx.x * blockDim.x + threadIdx.x; i < total; i += gridDim.x * blockDim.x) {
    int m = i / Kd, k = i - m * Kd;
    float v = (m < Mreal && k < Ks) ? src[m * Ks + k] : 0.f;
    dst[i] = f2bf(v);
  }
}

// x0[n,0:108]=node_features, [108:126]=node_config, [126:190]=opcode_table[ops[n]], [190:192]=0
__global__ void build_x0_kernel(const float* __restrict__ nf, const float* __restrict__ ncf,
                                const float* __restrict__ opt, const int* __restrict__ ops,
                                unsigned short* __restrict__ x0, int N) {
  int total = N * 192;
  for (int i = blockIdx.x * blockDim.x + threadIdx.x; i < total; i += gridDim.x * blockDim.x) {
    int row = i / 192, k = i - row * 192;
    float v;
    if (k < 108)       v = nf[(size_t)row * 108 + k];
    else if (k < 126)  v = ncf[(size_t)row * 18 + (k - 108)];
    else if (k < 190)  v = opt[(size_t)ops[row] * 64 + (k - 126)];
    else               v = 0.f;
    x0[i] = f2bf(v);
  }
}

// ---------------------------------------------------------------------------
// Generic bf16 WMMA GEMM:  out = leaky(A[N,K] @ W[M,K]^T + bias)
// 8 waves/block, wave strip = 16 rows x 64 cols, grid = (ceil(N/128), M/64).
// Optional fused per-segment sum / sumsq atomics (GraphwiseLayerNorm stats).
// ---------------------------------------------------------------------------
__global__ __launch_bounds__(256)
void gemm_bf16_kernel(const unsigned short* __restrict__ A,
                      const unsigned short* __restrict__ W,
                      const float* __restrict__ bias, int Mreal,
                      unsigned short* __restrict__ out_bf,
                      float* __restrict__ out_f32,
                      float* __restrict__ segS1, float* __restrict__ segS2,
                      const int* __restrict__ node_sep, int nseg, int maxStatCol,
                      int N, int K, int M) {
  const int lane = threadIdx.x & 31;
  const int wave = threadIdx.x >> 5;
  const int rowBase = blockIdx.x * 128 + wave * 16;
  const int colBase = blockIdx.y * 64;
  int arow = rowBase + (lane & 15);
  if (arow >= N) arow = N - 1;                 // clamp loads; stores are guarded
  const int alo = (lane >> 4) << 3;            // A-frag K sub-offset: lanes 16..31 -> +8
  const int blo = (lane >> 4) << 3;            // B-frag K sub-offset: lanes 16..31 -> +8
  const int mcol = lane & 15;

  v8f acc[4];
#pragma unroll
  for (int t = 0; t < 4; ++t)
#pragma unroll
    for (int i = 0; i < 8; ++i) acc[t][i] = 0.f;

  for (int kt = 0; kt < K; kt += 32) {
    Frag16 a;
    const unsigned short* ap = A + (size_t)arow * K + kt + alo;
    a.q[0] = *(const v4u*)ap;                  // K = base..base+7
    a.q[1] = *(const v4u*)(ap + 16);           // K = base+16..base+23
#pragma unroll
    for (int t = 0; t < 4; ++t) {
      const int m = colBase + t * 16 + mcol;
      Frag16 b;
      const unsigned short* bp = W + (size_t)m * K + kt + blo;
      b.q[0] = *(const v4u*)bp;                // K = base..base+7
      b.q[1] = *(const v4u*)(bp + 16);
      acc[t] = __builtin_amdgcn_wmma_f32_16x16x32_bf16(false, a.v, false, b.v,
                                                       (short)0, acc[t], false, false);
    }
  }

  const int rh = (lane >> 4) << 3;             // C/D layout: row = v + 8*(lane>=16)
#pragma unroll
  for (int v = 0; v < 8; ++v) {
    const int row = rowBase + rh + v;
    if (row >= N) continue;
    float s1 = 0.f, s2 = 0.f;
    bool anyStat = false;
#pragma unroll
    for (int t = 0; t < 4; ++t) {
      const int col = colBase + t * 16 + mcol;
      const float bb = (bias && col < Mreal) ? bias[col] : 0.f;
      float val = leaky(acc[t][v] + bb);
      if (out_bf)  out_bf[(size_t)row * M + col] = f2bf(val);
      if (out_f32) out_f32[(size_t)row * M + col] = val;
      if (segS1 && col < maxStatCol) { s1 += val; s2 += val * val; anyStat = true; }
    }
    if (segS1 && anyStat) {
      const int s = seg_of(row, node_sep, nseg);
      atomicAdd(&segS1[s], s1);
      atomicAdd(&segS2[s], s2);
    }
  }
}

// ---------------------------------------------------------------------------
// SAGE GEMM:  out = leaky(agg[N,256] @ L^T + x[N,256] @ R^T + bias), fp32 out
// ---------------------------------------------------------------------------
__global__ __launch_bounds__(256)
void sage_gemm_kernel(const unsigned short* __restrict__ Aagg,
                      const unsigned short* __restrict__ Ax,
                      const unsigned short* __restrict__ WL,
                      const unsigned short* __restrict__ WR,
                      const float* __restrict__ bias,
                      float* __restrict__ out, int N) {
  const int K = 256, M = 256;
  const int lane = threadIdx.x & 31;
  const int wave = threadIdx.x >> 5;
  const int rowBase = blockIdx.x * 128 + wave * 16;
  const int colBase = blockIdx.y * 64;
  int arow = rowBase + (lane & 15);
  if (arow >= N) arow = N - 1;
  const int alo = (lane >> 4) << 3;
  const int blo = (lane >> 4) << 3;
  const int mcol = lane & 15;

  v8f acc[4];
#pragma unroll
  for (int t = 0; t < 4; ++t)
#pragma unroll
    for (int i = 0; i < 8; ++i) acc[t][i] = 0.f;

  for (int kt = 0; kt < K; kt += 32) {
    Frag16 a1, a2;
    const unsigned short* p1 = Aagg + (size_t)arow * K + kt + alo;
    const unsigned short* p2 = Ax   + (size_t)arow * K + kt + alo;
    a1.q[0] = *(const v4u*)p1;  a1.q[1] = *(const v4u*)(p1 + 16);
    a2.q[0] = *(const v4u*)p2;  a2.q[1] = *(const v4u*)(p2 + 16);
#pragma unroll
    for (int t = 0; t < 4; ++t) {
      const int m = colBase + t * 16 + mcol;
      Frag16 bl, br;
      const unsigned short* q1 = WL + (size_t)m * K + kt + blo;
      const unsigned short* q2 = WR + (size_t)m * K + kt + blo;
      bl.q[0] = *(const v4u*)q1;  bl.q[1] = *(const v4u*)(q1 + 16);
      br.q[0] = *(const v4u*)q2;  br.q[1] = *(const v4u*)(q2 + 16);
      acc[t] = __builtin_amdgcn_wmma_f32_16x16x32_bf16(false, a1.v, false, bl.v,
                                                       (short)0, acc[t], false, false);
      acc[t] = __builtin_amdgcn_wmma_f32_16x16x32_bf16(false, a2.v, false, br.v,
                                                       (short)0, acc[t], false, false);
    }
  }

  const int rh = (lane >> 4) << 3;
#pragma unroll
  for (int v = 0; v < 8; ++v) {
    const int row = rowBase + rh + v;
    if (row >= N) continue;
#pragma unroll
    for (int t = 0; t < 4; ++t) {
      const int col = colBase + t * 16 + mcol;
      out[(size_t)row * M + col] = leaky(acc[t][v] + bias[col]);
    }
  }
}

// ---------------------------------------------------------------------------
// GraphwiseLayerNorm finalize + apply (in-place on bf16 activations)
// ---------------------------------------------------------------------------
__global__ void gln_finalize_kernel(const float* __restrict__ S1, const float* __restrict__ S2,
                                    const int* __restrict__ sep, float* __restrict__ mu,
                                    float* __restrict__ rinv, int nseg, int d) {
  int s = blockIdx.x * blockDim.x + threadIdx.x;
  if (s >= nseg) return;
  int start = s ? sep[s - 1] : 0;
  float cnt = (float)(sep[s] - start) * (float)d;
  float m = S1[s] / cnt;
  float var = S2[s] / cnt - m * m;
  mu[s] = m;
  rinv[s] = rsqrtf(var + 1e-5f);
}

__global__ void gln_apply_kernel(unsigned short* __restrict__ h, const float* __restrict__ mu,
                                 const float* __restrict__ rinv, const float* __restrict__ gw,
                                 const float* __restrict__ gb, const int* __restrict__ sep,
                                 int nseg, int N, int Mpad, int Mreal) {
  int total = N * Mpad;
  for (int i = blockIdx.x * blockDim.x + threadIdx.x; i < total; i += gridDim.x * blockDim.x) {
    int f = i % Mpad;
    if (f >= Mreal) continue;                  // keep zero padding
    int row = i / Mpad;
    int s = seg_of(row, sep, nseg);
    float v = bf2f(h[i]);
    v = (v - mu[s]) * rinv[s] * gw[f] + gb[f];
    h[i] = f2bf(v);
  }
}

// ---------------------------------------------------------------------------
// PairNorm: column sums + total sum-of-squares, finalize, apply
// pn[0..255] = colsum -> mu ; pn[256] = sum(x^2) ; pn[257] = scale
// ---------------------------------------------------------------------------
__global__ __launch_bounds__(256)
void pn_stats_kernel(const float* __restrict__ x, float* __restrict__ pn, int N) {
  const int f = threadIdx.x;
  float cs = 0.f, sq = 0.f;
  for (int row = blockIdx.x; row < N; row += gridDim.x) {
    float v = x[(size_t)row * 256 + f];
    cs += v; sq += v * v;
  }
  atomicAdd(&pn[f], cs);
  __shared__ float red[256];
  red[f] = sq; __syncthreads();
  for (int o = 128; o; o >>= 1) { if (f < o) red[f] += red[f + o]; __syncthreads(); }
  if (f == 0) atomicAdd(&pn[256], red[0]);
}

__global__ __launch_bounds__(256)
void pn_finalize_kernel(float* __restrict__ pn, int N) {
  const int f = threadIdx.x;
  float mu = pn[f] / (float)N;
  pn[f] = mu;
  __shared__ float red[256];
  red[f] = mu * mu; __syncthreads();
  for (int o = 128; o; o >>= 1) { if (f < o) red[f] += red[f + o]; __syncthreads(); }
  if (f == 0) {
    float msq = pn[256] / (float)N - red[0];   // mean over nodes of ||x-mu||^2
    pn[257] = 1.f / (1e-6f + sqrtf(fmaxf(msq, 0.f)));
  }
}

// x_norm (bf16) out; also zeroes agg buffer (fused)
__global__ void pn_apply_kernel(const float* __restrict__ x, const float* __restrict__ pn,
                                unsigned short* __restrict__ xnb, float* __restrict__ agg, int N) {
  int total = N * 256;
  float scale = pn[257];
  for (int i = blockIdx.x * blockDim.x + threadIdx.x; i < total; i += gridDim.x * blockDim.x) {
    int f = i & 255;
    float v = (x[i] - pn[f]) * scale;
    xnb[i] = f2bf(v);
    agg[i] = 0.f;
  }
}

// ---------------------------------------------------------------------------
// Edge degree + scatter-add (mean aggregation), agg -> bf16 with /deg
// ---------------------------------------------------------------------------
__global__ void deg_kernel(const int* __restrict__ dst, float* __restrict__ deg, int E) {
  for (int e = blockIdx.x * blockDim.x + threadIdx.x; e < E; e += gridDim.x * blockDim.x)
    atomicAdd(&deg[dst[e]], 1.f);
}

__global__ __launch_bounds__(256)
void scatter_kernel(const int* __restrict__ src, const int* __restrict__ dst,
                    const unsigned short* __restrict__ xnb, float* __restrict__ agg, int E) {
  int e = blockIdx.x * 4 + (threadIdx.x >> 6);  // 64 lanes per edge
  if (e >= E) return;
  const int s = src[e], d = dst[e];
  const int c0 = (threadIdx.x & 63) * 4;
#pragma unroll
  for (int j = 0; j < 4; ++j) {
    int c = c0 + j;
    atomicAdd(&agg[(size_t)d * 256 + c], bf2f(xnb[(size_t)s * 256 + c]));
  }
}

__global__ void aggb_kernel(const float* __restrict__ agg, const float* __restrict__ deg,
                            unsigned short* __restrict__ aggb, int N) {
  int total = N * 256;
  for (int i = blockIdx.x * blockDim.x + threadIdx.x; i < total; i += gridDim.x * blockDim.x) {
    int row = i >> 8;
    float d = deg[row]; d = d < 1.f ? 1.f : d;
    aggb[i] = f2bf(agg[i] / d);
  }
}

// ---------------------------------------------------------------------------
// Per-segment readout: out[seg] = sum_{n in seg} x[n,:] . wc + bc
// Segments are contiguous ranges (node_separation = cumulative ends).
// ---------------------------------------------------------------------------
__global__ __launch_bounds__(256)
void readout_kernel(const float* __restrict__ x, const float* __restrict__ wc,
                    const float* __restrict__ bc, const int* __restrict__ sep,
                    float* __restrict__ out, int nseg) {
  const int s = blockIdx.x;
  const int start = s ? sep[s - 1] : 0;
  const int end = sep[s];
  const int f = threadIdx.x;
  const float w = wc[f];
  float acc = 0.f;
  for (int n = start; n < end; ++n) acc += x[(size_t)n * 256 + f] * w;
  __shared__ float red[256];
  red[f] = acc; __syncthreads();
  for (int o = 128; o; o >>= 1) { if (f < o) red[f] += red[f + o]; __syncthreads(); }
  if (f == 0) out[s] = red[0] + bc[0];
}

// ---------------------------------------------------------------------------
// Host orchestration
// ---------------------------------------------------------------------------
extern "C" void kernel_launch(void* const* d_in, const int* in_sizes, int n_in,
                              void* d_out, int out_size, void* d_ws, size_t ws_size,
                              hipStream_t stream) {
  const float* nf  = (const float*)d_in[0];
  const float* ncf = (const float*)d_in[1];
  const float* opt = (const float*)d_in[2];
  const float* w1  = (const float*)d_in[3];
  const float* b1  = (const float*)d_in[4];
  const float* g1w = (const float*)d_in[5];
  const float* g1b = (const float*)d_in[6];
  const float* w2  = (const float*)d_in[7];
  const float* b2  = (const float*)d_in[8];
  const float* g2w = (const float*)d_in[9];
  const float* g2b = (const float*)d_in[10];
  const float* w3  = (const float*)d_in[11];
  const float* b3  = (const float*)d_in[12];
  const float* slw = (const float*)d_in[13];
  const float* slb = (const float*)d_in[14];
  const float* srw = (const float*)d_in[15];
  const float* wc  = (const float*)d_in[16];
  const float* bc  = (const float*)d_in[17];
  const int* sep   = (const int*)d_in[18];
  const int* ops   = (const int*)d_in[19];
  const int* edges = (const int*)d_in[20];
  // d_in[21] batches: sorted, identity mapping -> unused

  const int N = in_sizes[19];
  const int SEGn = in_sizes[18];
  const int E = in_sizes[20] / 2;
  const int* esrc = edges;
  const int* edst = edges + E;

  // bump allocator over workspace
  char* wp = (char*)d_ws;
  auto alloc = [&](size_t bytes) -> char* {
    char* p = wp; wp += (bytes + 255) & ~(size_t)255; return p;
  };
  unsigned short* x0   = (unsigned short*)alloc((size_t)N * 192 * 2);
  unsigned short* h1   = (unsigned short*)alloc((size_t)N * 384 * 2);
  unsigned short* h2   = (unsigned short*)alloc((size_t)N * 256 * 2);
  float*          xcur = (float*)         alloc((size_t)N * 256 * 4);
  unsigned short* xnb  = (unsigned short*)alloc((size_t)N * 256 * 2);
  float*          agg  = (float*)         alloc((size_t)N * 256 * 4);
  unsigned short* aggb = (unsigned short*)alloc((size_t)N * 256 * 2);
  float*          deg  = (float*)         alloc((size_t)N * 4);
  unsigned short* w1p  = (unsigned short*)alloc((size_t)384 * 192 * 2);
  unsigned short* w2p  = (unsigned short*)alloc((size_t)256 * 384 * 2);
  unsigned short* w3p  = (unsigned short*)alloc((size_t)256 * 256 * 2);
  unsigned short* slp  = (unsigned short*)alloc((size_t)3 * 256 * 256 * 2);
  unsigned short* srp  = (unsigned short*)alloc((size_t)3 * 256 * 256 * 2);
  float* segS  = (float*)alloc((size_t)2 * SEGn * 4);
  float* segS1 = segS;
  float* segS2 = segS + SEGn;
  float* mu    = (float*)alloc((size_t)SEGn * 4);
  float* rinv  = (float*)alloc((size_t)SEGn * 4);
  float* pn    = (float*)alloc(258 * 4);

  // --- weight conversion to padded bf16 ---
  conv_w_kernel<<<288, 256, 0, stream>>>(w1, w1p, 384, 380, 190, 192);
  conv_w_kernel<<<384, 256, 0, stream>>>(w2, w2p, 256, 256, 380, 384);
  conv_w_kernel<<<256, 256, 0, stream>>>(w3, w3p, 256, 256, 256, 256);
  for (int i = 0; i < 3; ++i) {
    conv_w_kernel<<<256, 256, 0, stream>>>(slw + (size_t)i * 65536, slp + (size_t)i * 65536,
                                           256, 256, 256, 256);
    conv_w_kernel<<<256, 256, 0, stream>>>(srw + (size_t)i * 65536, srp + (size_t)i * 65536,
                                           256, 256, 256, 256);
  }

  // --- input feature assembly (concat + opcode embed) ---
  build_x0_kernel<<<2048, 256, 0, stream>>>(nf, ncf, opt, ops, x0, N);

  const int gx = (N + 127) / 128;

  // --- layer 1: [N,192] x [384,192]^T, fused leaky + GLN stats ---
  zerof_kernel<<<2, 256, 0, stream>>>(segS, 2 * SEGn);
  gemm_bf16_kernel<<<dim3(gx, 6), 256, 0, stream>>>(x0, w1p, b1, 380, h1, nullptr,
                                                    segS1, segS2, sep, SEGn, 380, N, 192, 384);
  gln_finalize_kernel<<<1, 256, 0, stream>>>(segS1, segS2, sep, mu, rinv, SEGn, 380);
  gln_apply_kernel<<<4096, 256, 0, stream>>>(h1, mu, rinv, g1w, g1b, sep, SEGn, N, 384, 380);

  // --- layer 2: [N,384] x [256,384]^T ---
  zerof_kernel<<<2, 256, 0, stream>>>(segS, 2 * SEGn);
  gemm_bf16_kernel<<<dim3(gx, 4), 256, 0, stream>>>(h1, w2p, b2, 256, h2, nullptr,
                                                    segS1, segS2, sep, SEGn, 256, N, 384, 256);
  gln_finalize_kernel<<<1, 256, 0, stream>>>(segS1, segS2, sep, mu, rinv, SEGn, 256);
  gln_apply_kernel<<<4096, 256, 0, stream>>>(h2, mu, rinv, g2w, g2b, sep, SEGn, N, 256, 256);

  // --- layer 3: [N,256] x [256,256]^T -> fp32 x ---
  gemm_bf16_kernel<<<dim3(gx, 4), 256, 0, stream>>>(h2, w3p, b3, 256, nullptr, xcur,
                                                    nullptr, nullptr, sep, SEGn, 0, N, 256, 256);

  // --- degrees (once) ---
  zerof_kernel<<<1024, 256, 0, stream>>>(deg, N);
  deg_kernel<<<(E + 255) / 256, 256, 0, stream>>>(edst, deg, E);

  // --- 3 x (PairNorm -> SAGEConv) ---
  for (int i = 0; i < 3; ++i) {
    zerof_kernel<<<1, 256, 0, stream>>>(pn, 258);
    pn_stats_kernel<<<1024, 256, 0, stream>>>(xcur, pn, N);
    pn_finalize_kernel<<<1, 256, 0, stream>>>(pn, N);
    pn_apply_kernel<<<4096, 256, 0, stream>>>(xcur, pn, xnb, agg, N);
    scatter_kernel<<<(E + 3) / 4, 256, 0, stream>>>(esrc, edst, xnb, agg, E);
    aggb_kernel<<<4096, 256, 0, stream>>>(agg, deg, aggb, N);
    sage_gemm_kernel<<<dim3(gx, 4), 256, 0, stream>>>(aggb, xnb,
                                                      slp + (size_t)i * 65536,
                                                      srp + (size_t)i * 65536,
                                                      slb + (size_t)i * 256, xcur, N);
  }

  // --- per-segment readout fused with final linear ---
  readout_kernel<<<SEGn, 256, 0, stream>>>(xcur, wc, bc, sep, (float*)d_out, SEGn);
}
